// Attention_61357902791119
// MI455X (gfx1250) — matrix operand order repaired and checked by reading, not compiled, not run on hardware
//
#include <hip/hip_runtime.h>

// ---------------------------------------------------------------------------
// Multi-head attention (B=2, L=S=2048, d_model=1024, H=16, E=64) for gfx1250.
// All matmuls on v_wmma_f32_16x16x32_bf16 (wave32 WMMA, f32 accumulate).
// Conversion-free tile copies use CDNA5 async-to-LDS and the Tensor Data
// Mover (tensor_load_to_lds with LDS-padding descriptor).
// ---------------------------------------------------------------------------

#define D_MODEL 1024
#define N_HEADS 16
#define E_DIM   64
#define SEQ     2048
#define BATCH   2
#define ROWS    (BATCH * SEQ)   // 4096 tokens

typedef __attribute__((ext_vector_type(16))) __bf16 v16bf;
typedef __attribute__((ext_vector_type(8)))  float  v8f;
typedef __attribute__((ext_vector_type(4)))  unsigned int u32x4;
typedef __attribute__((ext_vector_type(8)))  unsigned int u32x8;

union FragU    { v16bf v; uint4 u[2]; };
union HalfPack { uint4 u; unsigned short h[8]; };

__device__ __forceinline__ unsigned short f2bf(float x) {
  unsigned u = __builtin_bit_cast(unsigned, x);
  u += 0x7FFFu + ((u >> 16) & 1u);          // round-to-nearest-even
  return (unsigned short)(u >> 16);
}
__device__ __forceinline__ unsigned pack2(float a, float b) {
  return (unsigned)f2bf(a) | ((unsigned)f2bf(b) << 16);
}
__device__ __forceinline__ v8f wmma_bf16(v16bf a, v16bf b, v8f c) {
  // (neg_a, A, neg_b, B, c_mod, C, reuse_a, reuse_b)
  return __builtin_amdgcn_wmma_f32_16x16x32_bf16(false, a, false, b,
                                                 (short)0, c, false, false);
}

// ---- CDNA5 async global->LDS copy (ASYNCcnt) ------------------------------
// LDS operand = low 32 bits of the generic shared pointer (ISA 10.2: the LDS
// aperture maps addr[31:0] directly onto the LDS byte offset).
__device__ __forceinline__ unsigned lds_off(const void* p) {
  return (unsigned)(unsigned long long)p;
}
__device__ __forceinline__ void async_g2l_b128(unsigned loff, const void* gsrc) {
  asm volatile("global_load_async_to_lds_b128 %0, %1, off"
               :: "v"(loff), "v"(gsrc) : "memory");
}
__device__ __forceinline__ void wait_async0() {
  asm volatile("s_wait_asynccnt 0" ::: "memory");
}

// ---- CDNA5 Tensor Data Mover: 2D bf16 tile load with LDS padding ----------
// Loads a (tileH x tileW)-element 2-byte tile from a row-major tensor with
// row stride `strideElems` into LDS at `loff`, inserting `padDwords` of LDS
// padding after every 32 dwords (= one 64-elem row), i.e. 144-B LDS rows.
__device__ __forceinline__ void tdm_load_tile_64x64_pad(unsigned loff,
                                                        const void* gtile,
                                                        unsigned tensorH) {
  unsigned long long ga = (unsigned long long)gtile;
  u32x4 g0;
  g0[0] = 1u;                                        // count=1 user descriptor
  g0[1] = loff;                                      // lds_addr
  g0[2] = (unsigned)(ga & 0xFFFFFFFFu);              // global_addr[31:0]
  g0[3] = (unsigned)((ga >> 32) & 0x01FFFFFFu)       // global_addr[56:32]
        | (2u << 30);                                // type = 2 ("image")
  u32x8 g1;
  g1[0] = (1u << 16)        // data_size = 1 (2 bytes)
        | (1u << 20)        // pad_enable
        | (4u << 22)        // pad_interval: 32 dwords (one 64-elem row)
        | (3u << 25);       // pad_amount:   4 dwords (16 B) -> 144 B rows
  g1[1] = (64u << 16);                 // tensor_dim0 = 64 (bits 79:48, low)
  g1[2] = (tensorH & 0xFFFFu) << 16;   // tensor_dim1 (bits 111:80, low)
  g1[3] = (tensorH >> 16) | (64u << 16); // tensor_dim1 hi | tile_dim0 = 64
  g1[4] = 64u;                         // tile_dim1 = 64, tile_dim2 = 0
  g1[5] = 64u;                         // tensor_dim0_stride = 64 elems
  g1[6] = 0u;
  g1[7] = 0u;
  asm volatile("tensor_load_to_lds %0, %1" :: "s"(g0), "s"(g1) : "memory");
}

// ---------------------------------------------------------------------------
// Tiled GEMM: C[4096 x 1024] = A[4096 x 1024] * W[1024 x 1024] + bias
//  - 128x128 block tile, BK=32 (one WMMA K-step), 8 waves: 4x2 wave grid,
//    each wave owns 32x64 (2x4 accumulators of 16x16).
// ---------------------------------------------------------------------------
template<bool A_BF16, bool OUT_HEADS>
__global__ __launch_bounds__(256)
void gemm_wmma(const float* __restrict__ Af32,
               const unsigned short* __restrict__ Abf,
               const float* __restrict__ W,
               const float* __restrict__ bias,
               unsigned short* __restrict__ outBf,
               float* __restrict__ outF) {
  __shared__ unsigned short As[128][40];   // [m][k], 80 B row stride (16B mult)
  __shared__ unsigned short Bs[128][40];   // transposed: [n][k]

  const int t    = threadIdx.x;
  const int lane = t & 31;
  const int wave = t >> 5;
  const int hlf  = lane >> 4;    // which 16-lane half
  const int lm   = lane & 15;
  const int waveRow = (wave >> 1) * 32;    // 0,32,64,96
  const int waveCol = (wave & 1)  * 64;    // 0,64
  const int rowBase = blockIdx.y * 128;
  const int colBase = blockIdx.x * 128;

  v8f acc[2][4] = {};

  const int ar = t >> 1;             // A staging: row 0..127
  const int ak = (t & 1) * 16;       //            k half 0/16
  const int bk = t >> 3;             // B staging: k 0..31
  const int bn = (t & 7) * 16;       //            n chunk

  for (int k0 = 0; k0 < D_MODEL; k0 += 32) {
    __syncthreads();

    // ---- stage A tile ----
    if constexpr (A_BF16) {
      // already bf16: pure copy -> async global->LDS path (ASYNCcnt)
      const unsigned short* src = Abf + (size_t)(rowBase + ar) * D_MODEL + k0 + ak;
      const unsigned loff = lds_off(&As[ar][ak]);
      async_g2l_b128(loff,      src);
      async_g2l_b128(loff + 16, src + 8);
      if (k0 + 32 < D_MODEL) __builtin_prefetch(src + 32, 0, 1);
    } else {
      // f32 -> bf16 conversion while staging
      const float* src = Af32 + (size_t)(rowBase + ar) * D_MODEL + k0 + ak;
      float4 f0 = ((const float4*)src)[0];
      float4 f1 = ((const float4*)src)[1];
      float4 f2 = ((const float4*)src)[2];
      float4 f3 = ((const float4*)src)[3];
      uint4 lo = { pack2(f0.x, f0.y), pack2(f0.z, f0.w),
                   pack2(f1.x, f1.y), pack2(f1.z, f1.w) };
      uint4 hi = { pack2(f2.x, f2.y), pack2(f2.z, f2.w),
                   pack2(f3.x, f3.y), pack2(f3.z, f3.w) };
      *(uint4*)&As[ar][ak]     = lo;
      *(uint4*)&As[ar][ak + 8] = hi;
      if (k0 + 32 < D_MODEL) __builtin_prefetch(src + 32, 0, 1);
    }

    // ---- stage B tile transposed: Bs[n][k] = W[(k0+k)*N + colBase + n] ----
    {
      const float* src = W + (size_t)(k0 + bk) * D_MODEL + colBase + bn;
      if (k0 + 32 < D_MODEL) __builtin_prefetch(src + 32 * D_MODEL, 0, 1);
#pragma unroll
      for (int j = 0; j < 16; j += 4) {
        float4 f = *(const float4*)(src + j);
        Bs[bn + j + 0][bk] = f2bf(f.x);
        Bs[bn + j + 1][bk] = f2bf(f.y);
        Bs[bn + j + 2][bk] = f2bf(f.z);
        Bs[bn + j + 3][bk] = f2bf(f.w);
      }
    }
    if constexpr (A_BF16) wait_async0();
    __syncthreads();

    // ---- fragments + 2x4 WMMA ----
    FragU aF[2], bF[4];
#pragma unroll
    for (int i = 0; i < 2; ++i) {
      const unsigned short* r = &As[waveRow + i * 16 + lm][0];
      aF[i].u[0] = *(const uint4*)(r + hlf * 16);       // K = half*16 .. +7
      aF[i].u[1] = *(const uint4*)(r + hlf * 16 + 8);   // K = half*16+8 .. +15
    }
#pragma unroll
    for (int j = 0; j < 4; ++j) {
      const unsigned short* r = &Bs[waveCol + j * 16 + lm][0];
      bF[j].u[0] = *(const uint4*)(r + hlf * 16);       // K = half*16 .. +7
      bF[j].u[1] = *(const uint4*)(r + hlf * 16 + 8);   // K = half*16+8 .. +15
    }
#pragma unroll
    for (int i = 0; i < 2; ++i)
#pragma unroll
      for (int j = 0; j < 4; ++j)
        acc[i][j] = wmma_bf16(aF[i].v, bF[j].v, acc[i][j]);
  }

  // ---- epilogue ----
#pragma unroll
  for (int j = 0; j < 4; ++j) {
    const int nG = colBase + waveCol + j * 16 + lm;   // column (fixed per lane)
    const float bv = bias[nG];
#pragma unroll
    for (int i = 0; i < 2; ++i) {
#pragma unroll
      for (int r = 0; r < 8; ++r) {
        const int mG = rowBase + waveRow + i * 16 + hlf * 8 + r;  // C/D layout
        const float val = acc[i][j][r] + bv;
        if constexpr (OUT_HEADS) {
          const int h = nG >> 6, e = nG & 63;
          const int b = mG >> 11, l = mG & 2047;
          outBf[(((size_t)(b * N_HEADS + h)) * SEQ + l) * E_DIM + e] = f2bf(val);
        } else {
          outF[(size_t)mG * D_MODEL + nG] = val;
        }
      }
    }
  }
}

// ---------------------------------------------------------------------------
// Flash attention over bf16 Q/K/V in [B*H, SEQ, E] layout.
// One block (4 waves) per 64 query rows of one (b,h). Each wave owns 16 rows.
// Q tile: async global->LDS.  K tile: TDM tensor_load_to_lds (padded rows).
// V tile: VALU transpose into LDS.  Online softmax in f32.
// ---------------------------------------------------------------------------
__global__ __launch_bounds__(128)
void attn_wmma(const unsigned short* __restrict__ qb,
               const unsigned short* __restrict__ kb,
               const unsigned short* __restrict__ vb,
               unsigned short* __restrict__ ob) {
  __shared__ unsigned short Qs[64][72];    // [row][e]   144 B stride
  __shared__ unsigned short Ks[64][72];    // [s][e]  == B^T layout for QK^T
  __shared__ unsigned short VTs[64][72];   // [e][s]  == B^T layout for P*V
  __shared__ unsigned short Ps[64][72];    // [row][s]

  const int t    = threadIdx.x;
  const int lane = t & 31;
  const int wave = t >> 5;
  const int hlf  = lane >> 4;
  const int lm   = lane & 15;
  const int rowW = wave * 16;

  const int bh = blockIdx.x;                 // b*H + h
  const int b = bh >> 4, h = bh & 15;
  const size_t headOff = (size_t)bh * SEQ * E_DIM;
  const int qRow0 = blockIdx.y * 64;

  const int lr = t >> 1;            // staging row 0..63
  const int lp = (t & 1) * 32;      // e-half 0/32

  // ---- resident Q tile via async global->LDS ----
  {
    const unsigned short* src = qb + headOff + (size_t)(qRow0 + lr) * E_DIM + lp;
    const unsigned loff = lds_off(&Qs[lr][lp]);
#pragma unroll
    for (int q4 = 0; q4 < 4; ++q4)
      async_g2l_b128(loff + q4 * 16, src + q4 * 8);
  }

  float mstat[8], lstat[8];
#pragma unroll
  for (int r = 0; r < 8; ++r) { mstat[r] = -3.0e38f; lstat[r] = 0.f; }
  v8f oacc[4] = {};

  for (int s0 = 0; s0 < SEQ; s0 += 64) {
    __syncthreads();
    // ---- K tile [s][e] via Tensor Data Mover (one descriptor per block) ----
    if (wave == 0) {
      tdm_load_tile_64x64_pad(lds_off(&Ks[0][0]),
                              kb + headOff + (size_t)s0 * E_DIM, SEQ);
    }
    // ---- V tile transposed [e][s] ----
    {
      const uint4* src = (const uint4*)(vb + headOff + (size_t)(s0 + lr) * E_DIM + lp);
#pragma unroll
      for (int q4 = 0; q4 < 4; ++q4) {
        HalfPack hp; hp.u = src[q4];
#pragma unroll
        for (int jj = 0; jj < 8; ++jj)
          VTs[lp + q4 * 8 + jj][lr] = hp.h[jj];
      }
    }
    wait_async0();                                   // Q tile (iter 0)
    if (wave == 0) __builtin_amdgcn_s_wait_tensorcnt(0);  // K tile
    __syncthreads();

    // ---- S = Q K^T  (16x64 per wave, 8 WMMA) ----
    v8f sacc[4] = {};
#pragma unroll
    for (int e0 = 0; e0 < 64; e0 += 32) {
      FragU a;
      const unsigned short* qr = &Qs[rowW + lm][0];
      a.u[0] = *(const uint4*)(qr + e0 + hlf * 16);
      a.u[1] = *(const uint4*)(qr + e0 + hlf * 16 + 8);
#pragma unroll
      for (int j = 0; j < 4; ++j) {
        FragU bf;
        const unsigned short* kr = &Ks[j * 16 + lm][0];
        bf.u[0] = *(const uint4*)(kr + e0 + hlf * 16);
        bf.u[1] = *(const uint4*)(kr + e0 + hlf * 16 + 8);
        sacc[j] = wmma_bf16(a.v, bf.v, sacc[j]);
      }
    }

    // ---- online softmax; cross-lane reduce within 16-lane row groups ----
#pragma unroll
    for (int r = 0; r < 8; ++r) {
      const float x0 = sacc[0][r] * 0.125f;   // 1/sqrt(E)
      const float x1 = sacc[1][r] * 0.125f;
      const float x2 = sacc[2][r] * 0.125f;
      const float x3 = sacc[3][r] * 0.125f;
      float rmax = fmaxf(fmaxf(x0, x1), fmaxf(x2, x3));
#pragma unroll
      for (int m = 1; m <= 8; m <<= 1)
        rmax = fmaxf(rmax, __shfl_xor(rmax, m, 32));
      const float mnew  = fmaxf(mstat[r], rmax);
      const float alpha = __expf(mstat[r] - mnew);
      mstat[r] = mnew;
      const float p0 = __expf(x0 - mnew);
      const float p1 = __expf(x1 - mnew);
      const float p2 = __expf(x2 - mnew);
      const float p3 = __expf(x3 - mnew);
      float rs = (p0 + p1) + (p2 + p3);
#pragma unroll
      for (int m = 1; m <= 8; m <<= 1)
        rs += __shfl_xor(rs, m, 32);
      lstat[r] = lstat[r] * alpha + rs;
#pragma unroll
      for (int j = 0; j < 4; ++j) oacc[j][r] *= alpha;
      const int prow = rowW + hlf * 8 + r;    // C-layout row -> LDS
      Ps[prow][ 0 + lm] = f2bf(p0);
      Ps[prow][16 + lm] = f2bf(p1);
      Ps[prow][32 + lm] = f2bf(p2);
      Ps[prow][48 + lm] = f2bf(p3);
    }
    __syncthreads();

    // ---- O += P V  (8 WMMA) ----
#pragma unroll
    for (int k0 = 0; k0 < 64; k0 += 32) {
      FragU a;
      const unsigned short* pr = &Ps[rowW + lm][0];
      a.u[0] = *(const uint4*)(pr + k0 + hlf * 16);
      a.u[1] = *(const uint4*)(pr + k0 + hlf * 16 + 8);
#pragma unroll
      for (int j = 0; j < 4; ++j) {
        FragU bf;
        const unsigned short* vr = &VTs[j * 16 + lm][0];
        bf.u[0] = *(const uint4*)(vr + k0 + hlf * 16);
        bf.u[1] = *(const uint4*)(vr + k0 + hlf * 16 + 8);
        oacc[j] = wmma_bf16(a.v, bf.v, oacc[j]);
      }
    }
  }

  // ---- normalize + write bf16 O into token-major [B*SEQ, D_MODEL] ----
#pragma unroll
  for (int j = 0; j < 4; ++j) {
    const int col = h * E_DIM + j * 16 + lm;
#pragma unroll
    for (int r = 0; r < 8; ++r) {
      const int lrow = qRow0 + rowW + hlf * 8 + r;
      const float o = oacc[j][r] / lstat[r];
      ob[(size_t)(b * SEQ + lrow) * D_MODEL + col] = f2bf(o);
    }
  }
}

// ---------------------------------------------------------------------------
extern "C" void kernel_launch(void* const* d_in, const int* in_sizes, int n_in,
                              void* d_out, int out_size, void* d_ws, size_t ws_size,
                              hipStream_t stream) {
  const float* queries = (const float*)d_in[0];
  const float* keys    = (const float*)d_in[1];
  const float* values  = (const float*)d_in[2];
  const float* Wq = (const float*)d_in[3];  const float* bq = (const float*)d_in[4];
  const float* Wk = (const float*)d_in[5];  const float* bk = (const float*)d_in[6];
  const float* Wv = (const float*)d_in[7];  const float* bv = (const float*)d_in[8];
  const float* Wo = (const float*)d_in[9];  const float* bo = (const float*)d_in[10];
  float* out = (float*)d_out;

  // workspace: 4 x (4096*1024) bf16 buffers = 32 MB
  unsigned short* qbuf = (unsigned short*)d_ws;
  unsigned short* kbuf = qbuf + (size_t)ROWS * D_MODEL;
  unsigned short* vbuf = kbuf + (size_t)ROWS * D_MODEL;
  unsigned short* obuf = vbuf + (size_t)ROWS * D_MODEL;

  dim3 gGrid(D_MODEL / 128, ROWS / 128);   // (8, 32)

  gemm_wmma<false, true><<<gGrid, 256, 0, stream>>>(queries, nullptr, Wq, bq, qbuf, nullptr);
  gemm_wmma<false, true><<<gGrid, 256, 0, stream>>>(keys,    nullptr, Wk, bk, kbuf, nullptr);
  gemm_wmma<false, true><<<gGrid, 256, 0, stream>>>(values,  nullptr, Wv, bv, vbuf, nullptr);

  dim3 aGrid(BATCH * N_HEADS, SEQ / 64);   // (32, 32)
  attn_wmma<<<aGrid, 128, 0, stream>>>(qbuf, kbuf, vbuf, obuf);

  gemm_wmma<true, false><<<gGrid, 256, 0, stream>>>(nullptr, obuf, Wo, bo, nullptr, out);
}
